// TransformerBlock_51694226375123
// MI455X (gfx1250) — compile-verified, matrix operand order checked by reading, not run on hardware
//
#include <hip/hip_runtime.h>

// ---------------------------------------------------------------------------
// Transformer block forward for MI455X (gfx1250), bf16 WMMA + f32 accumulate.
// x:[2,2048,1024] fp32.  All GEMMs via v_wmma_f32_16x16x32_bf16.
// Staging uses GLOBAL_LOAD_ASYNC_TO_LDS (ASYNCcnt) when the builtin exists,
// with a VGPR-mediated fallback; GEMM K-loop is LDS double-buffered.
// ---------------------------------------------------------------------------

typedef __attribute__((ext_vector_type(16))) __bf16 v16bf;
typedef __attribute__((ext_vector_type(8)))  __bf16 v8bf;
typedef __attribute__((ext_vector_type(8)))  float  v8f;

#define WIDTH_C   1024
#define SEQ_N     2048
#define BATCH_B   2
#define HEADS_H   16
#define HEAD_D    64
#define HIDDEN_F  4096
#define ROWS_M    (BATCH_B * SEQ_N)       // 4096

#if defined(__AMDGCN__) && __has_builtin(__builtin_amdgcn_global_load_async_to_lds_b128) && __has_builtin(__builtin_amdgcn_s_wait_asynccnt)
#define HAVE_ASYNC_LDS 1
#else
#define HAVE_ASYNC_LDS 0
#endif

static __device__ __forceinline__ __bf16 f2bf(float f) {
    unsigned u; __builtin_memcpy(&u, &f, 4);
    u += 0x7FFFu + ((u >> 16) & 1u);       // round-to-nearest-even
    unsigned short h = (unsigned short)(u >> 16);
    __bf16 r; __builtin_memcpy(&r, &h, 2);
    return r;
}

// 16-byte global -> LDS copy; async (ASYNCcnt) when available.
// Builtin signature (from compiler diagnostic): param 1 is
// '__device__ int __attribute__((vector_size(16)))*' (global src),
// param 2 the LDS dst, then imm offset + imm cpol.
#if HAVE_ASYNC_LDS
typedef int v4i_vs __attribute__((vector_size(16)));
static __device__ __forceinline__ void copy16_to_lds(const __bf16* g, __bf16* l) {
    __builtin_amdgcn_global_load_async_to_lds_b128(
        (__attribute__((address_space(1))) v4i_vs*)(g),
        (__attribute__((address_space(3))) v4i_vs*)(l), 0, 0);
}
static __device__ __forceinline__ void async_wait0() {
    __builtin_amdgcn_s_wait_asynccnt(0);
}
#else
static __device__ __forceinline__ void copy16_to_lds(const __bf16* g, __bf16* l) {
    *(v8bf*)l = *(const v8bf*)g;
}
static __device__ __forceinline__ void async_wait0() {}
#endif

static __device__ __forceinline__ v16bf cat16(v8bf lo, v8bf hi) {
    return __builtin_shufflevector(lo, hi, 0,1,2,3,4,5,6,7,8,9,10,11,12,13,14,15);
}

// Load a 16-bit A/B WMMA fragment from a K-contiguous 32-element row.
// Per ISA: lane half 0 holds K = {0..7, 16..23}; half 1 holds K = {8..15, 24..31}.
static __device__ __forceinline__ v16bf load_frag_row(const __bf16* row, int half) {
    v8bf lo = *(const v8bf*)(row + 8 * half);
    v8bf hi = *(const v8bf*)(row + 16 + 8 * half);
    return cat16(lo, hi);
}

// ---------------------------------------------------------------------------
// fp32 -> bf16 cast (weights)
// ---------------------------------------------------------------------------
__global__ __launch_bounds__(256)
void cast_f32_bf16(const float* __restrict__ in, __bf16* __restrict__ out, int n) {
    int i = blockIdx.x * 256 + threadIdx.x;
    if (i < n) out[i] = f2bf(in[i]);
}

// ---------------------------------------------------------------------------
// LayerNorm over last dim (1024), output bf16. One block (256 thr) per row.
// ---------------------------------------------------------------------------
__global__ __launch_bounds__(256)
void layernorm_bf16(const float* __restrict__ x, const float* __restrict__ g,
                    const float* __restrict__ bta, __bf16* __restrict__ out) {
    __shared__ float s1[256];
    __shared__ float s2[256];
    const int row = blockIdx.x;
    const int tid = threadIdx.x;
    const float* xr = x + (size_t)row * WIDTH_C;
    float v[4], sum = 0.f, sq = 0.f;
#pragma unroll
    for (int i = 0; i < 4; ++i) {
        v[i] = xr[tid + 256 * i];
        sum += v[i];
        sq  += v[i] * v[i];
    }
    s1[tid] = sum; s2[tid] = sq;
    __syncthreads();
    for (int off = 128; off > 0; off >>= 1) {
        if (tid < off) { s1[tid] += s1[tid + off]; s2[tid] += s2[tid + off]; }
        __syncthreads();
    }
    const float mu   = s1[0] * (1.f / WIDTH_C);
    const float var  = s2[0] * (1.f / WIDTH_C) - mu * mu;
    const float rstd = rsqrtf(var + 1e-5f);
#pragma unroll
    for (int i = 0; i < 4; ++i) {
        int c = tid + 256 * i;
        out[(size_t)row * WIDTH_C + c] = f2bf((v[i] - mu) * rstd * g[c] + bta[c]);
    }
}

// ---------------------------------------------------------------------------
// Tiled bf16 GEMM: C[M,N] = A[M,K] * B[K,N]  (+ epilogue), WMMA 16x16x32.
// Block: 256 threads = 8 waves. Block tile 128x128, K-step 32, LDS
// double-buffered: slab k+32 is staged (async-to-LDS for A) while the 8
// WMMAs per wave consume slab k.  Wave tile 32x64 -> 2x4 v8f accumulators.
// EPI 0: QKV scatter -> q/k/v bf16 [B,H,N,D]
// EPI 1: out f32 = acc + bias[n] + res[r,n]
// EPI 2: out bf16 = exact GELU(acc + bias[n])
// M,N,K all multiples of tile sizes for this problem (no bounds checks).
// ---------------------------------------------------------------------------
template <int EPI>
__global__ __launch_bounds__(256)
void gemm_bf16_wmma(const __bf16* __restrict__ A, const __bf16* __restrict__ Bw,
                    int M, int N, int Kd,
                    const float* __restrict__ bias, const float* __restrict__ res,
                    float* __restrict__ outF, __bf16* __restrict__ outB,
                    __bf16* __restrict__ qd, __bf16* __restrict__ kd,
                    __bf16* __restrict__ vd) {
    __shared__ __align__(16) __bf16 Asl[2][128][32];   // [buf][m][k]  K-contiguous
    __shared__ __align__(16) __bf16 Bsl[2][128][32];   // [buf][n][k]  transposed

    const int tid  = threadIdx.x;
    const int lane = tid & 31;
    const int wave = tid >> 5;
    const int half = lane >> 4;
    const int l16  = lane & 15;
    const int wm   = wave & 3;   // 4 waves along M (4*32 = 128)
    const int wn   = wave >> 2;  // 2 waves along N (2*64 = 128)
    const int m0   = blockIdx.y * 128;
    const int n0   = blockIdx.x * 128;

    // staging coordinates (2 x 16B chunks per thread per tile)
    const int ar0 = tid >> 2,          akc0 = (tid & 3) << 3;          // A chunk 0
    const int ar1 = (tid + 256) >> 2,  akc1 = ((tid + 256) & 3) << 3;  // A chunk 1
    const int bk0 = tid >> 4,          bnc0 = (tid & 15) << 3;         // B chunk 0
    const int bk1 = (tid + 256) >> 4,  bnc1 = ((tid + 256) & 15) << 3; // B chunk 1

    auto stage = [&](int buf, int k0) {
        // A tile (128x32): contiguous 16B -> async copy straight to LDS
        const __bf16* sa0 = A + (size_t)(m0 + ar0) * Kd + k0 + akc0;
        const __bf16* sa1 = A + (size_t)(m0 + ar1) * Kd + k0 + akc1;
        copy16_to_lds(sa0, &Asl[buf][ar0][akc0]);
        copy16_to_lds(sa1, &Asl[buf][ar1][akc1]);
        __builtin_prefetch(sa0 + 64, 0, 1);
        // B tile (32x128) transposed into [n][k]: VGPR-mediated
        const __bf16* sb0 = Bw + (size_t)(k0 + bk0) * N + n0 + bnc0;
        const __bf16* sb1 = Bw + (size_t)(k0 + bk1) * N + n0 + bnc1;
        v8bf vb0 = *(const v8bf*)sb0;
        v8bf vb1 = *(const v8bf*)sb1;
        __builtin_prefetch(sb0 + (size_t)32 * N, 0, 1);
#pragma unroll
        for (int e = 0; e < 8; ++e) {
            Bsl[buf][bnc0 + e][bk0] = vb0[e];
            Bsl[buf][bnc1 + e][bk1] = vb1[e];
        }
    };

    v8f acc[2][4];
#pragma unroll
    for (int mt = 0; mt < 2; ++mt)
#pragma unroll
        for (int nt = 0; nt < 4; ++nt)
#pragma unroll
            for (int i = 0; i < 8; ++i) acc[mt][nt][i] = 0.f;

    stage(0, 0);
    async_wait0();
    __syncthreads();

    int cur = 0;
    for (int k0 = 0; k0 < Kd; k0 += 32) {
        const int nxt = cur ^ 1;
        if (k0 + 32 < Kd) stage(nxt, k0 + 32);   // overlaps with WMMAs below

        v16bf afr[2], bfr[4];
#pragma unroll
        for (int mt = 0; mt < 2; ++mt)
            afr[mt] = load_frag_row(Asl[cur][wm * 32 + mt * 16 + l16], half);
#pragma unroll
        for (int nt = 0; nt < 4; ++nt)
            bfr[nt] = load_frag_row(Bsl[cur][wn * 64 + nt * 16 + l16], half);

#pragma unroll
        for (int mt = 0; mt < 2; ++mt)
#pragma unroll
            for (int nt = 0; nt < 4; ++nt)
                acc[mt][nt] = __builtin_amdgcn_wmma_f32_16x16x32_bf16(
                    false, afr[mt], false, bfr[nt], (short)0, acc[mt][nt],
                    false, false);

        async_wait0();       // next-slab async A copies landed in LDS
        __syncthreads();     // + everyone's ds stores visible / reads done
        cur = nxt;
    }

    // ----- epilogue (C/D layout: VGPR v -> row v+8*half, col = lane%16) -----
#pragma unroll
    for (int mt = 0; mt < 2; ++mt)
#pragma unroll
        for (int nt = 0; nt < 4; ++nt)
#pragma unroll
            for (int v = 0; v < 8; ++v) {
                const int r  = m0 + wm * 32 + mt * 16 + v + 8 * half;
                const int cN = n0 + wn * 64 + nt * 16 + l16;
                float val = acc[mt][nt][v];
                if (EPI == 0) {
                    // scatter [M, 3C] -> q/k/v [B,H,N,D]
                    int part = cN >> 10, within = cN & 1023;
                    int head = within >> 6, d = within & 63;
                    int bb = r >> 11, n = r & 2047;
                    size_t idx =
                        (((size_t)(bb * HEADS_H + head)) * SEQ_N + n) * HEAD_D + d;
                    __bf16 bv = f2bf(val);
                    (part == 0 ? qd : part == 1 ? kd : vd)[idx] = bv;
                } else if (EPI == 1) {
                    size_t idx = (size_t)r * N + cN;
                    outF[idx] = val + bias[cN] + res[idx];
                } else {
                    float t  = val + bias[cN];
                    float gl = 0.5f * t * (1.0f + erff(t * 0.70710678118654752f));
                    outB[(size_t)r * N + cN] = f2bf(gl);
                }
            }
}

// ---------------------------------------------------------------------------
// Flash attention, bf16 WMMA, fp32 online softmax.
// grid (N/128, H, B), 256 threads = 8 waves; each wave owns 16 query rows.
// Block stages 32-key K/V tiles in LDS (K tile via async-to-LDS);
// per step: 4 WMMA (S) + 4 WMMA (PV).
// ---------------------------------------------------------------------------
__global__ __launch_bounds__(256)
void attention_wmma(const __bf16* __restrict__ Q, const __bf16* __restrict__ Kt,
                    const __bf16* __restrict__ Vt, __bf16* __restrict__ Out) {
    __shared__ __align__(16) __bf16 Ksl[32][64];      // [key][d]
    __shared__ __align__(16) __bf16 Vsl[64][32];      // [d][key]  (transposed)
    __shared__ __align__(16) __bf16 Psl[8][16][32];   // per-wave P bounce

    const int tid  = threadIdx.x;
    const int lane = tid & 31;
    const int wave = tid >> 5;
    const int half = lane >> 4;
    const int l16  = lane & 15;
    const int h = blockIdx.y, b = blockIdx.z;
    const int qrow = blockIdx.x * 128 + wave * 16 + l16;
    const size_t bh = ((size_t)(b * HEADS_H + h)) * SEQ_N;

    // Q fragments for K-chunks 0..31 and 32..63 of the d dimension
    const __bf16* qp = Q + (bh + qrow) * HEAD_D;
    v16bf aq[2];
#pragma unroll
    for (int c = 0; c < 2; ++c) aq[c] = load_frag_row(qp + c * 32, half);

    float mrun[8], lrun[8];
    v8f o[4];
#pragma unroll
    for (int v = 0; v < 8; ++v) { mrun[v] = -3.0e38f; lrun[v] = 0.f; }
#pragma unroll
    for (int dt = 0; dt < 4; ++dt)
#pragma unroll
        for (int v = 0; v < 8; ++v) o[dt][v] = 0.f;

    const int kr = tid >> 3;              // key row 0..31 for staging
    const int dc = (tid & 7) << 3;        // d column chunk

    for (int j = 0; j < SEQ_N / 32; ++j) {
        // stage K (async straight to LDS) and V (transposed, VGPR-mediated)
        copy16_to_lds(Kt + (bh + (size_t)j * 32 + kr) * HEAD_D + dc,
                      &Ksl[kr][dc]);
        v8bf vv = *(const v8bf*)(Vt + (bh + (size_t)j * 32 + kr) * HEAD_D + dc);
#pragma unroll
        for (int e = 0; e < 8; ++e) Vsl[dc + e][kr] = vv[e];
        async_wait0();
        __syncthreads();

        // S = Q * K^T for two 16-key tiles; accumulate over d (two 32-chunks)
        v8f s[2];
#pragma unroll
        for (int t = 0; t < 2; ++t) {
#pragma unroll
            for (int i = 0; i < 8; ++i) s[t][i] = 0.f;
#pragma unroll
            for (int c = 0; c < 2; ++c) {
                v16bf kf = load_frag_row(&Ksl[t * 16 + l16][c * 32], half);
                s[t] = __builtin_amdgcn_wmma_f32_16x16x32_bf16(
                    false, aq[c], false, kf, (short)0, s[t], false, false);
            }
        }

        // online softmax; row = v + 8*half, reduce across the 16-lane half
        float corr[8];
#pragma unroll
        for (int v = 0; v < 8; ++v) {
            float s0 = s[0][v] * 0.125f;          // 1/sqrt(64)
            float s1 = s[1][v] * 0.125f;
            float rm = fmaxf(s0, s1);
            rm = fmaxf(rm, __shfl_xor(rm, 1));
            rm = fmaxf(rm, __shfl_xor(rm, 2));
            rm = fmaxf(rm, __shfl_xor(rm, 4));
            rm = fmaxf(rm, __shfl_xor(rm, 8));
            float mnew = fmaxf(mrun[v], rm);
            corr[v] = __expf(mrun[v] - mnew);
            mrun[v] = mnew;
            float p0 = __expf(s0 - mnew);
            float p1 = __expf(s1 - mnew);
            lrun[v] = lrun[v] * corr[v] + p0 + p1;
            Psl[wave][v + 8 * half][l16]      = f2bf(p0);
            Psl[wave][v + 8 * half][16 + l16] = f2bf(p1);
        }
#pragma unroll
        for (int dt = 0; dt < 4; ++dt)
#pragma unroll
            for (int v = 0; v < 8; ++v) o[dt][v] *= corr[v];

        // O += P(16x32) * V(32x64); P re-read in A-fragment layout
        v16bf pf = load_frag_row(Psl[wave][l16], half);
#pragma unroll
        for (int dt = 0; dt < 4; ++dt) {
            v16bf vf = load_frag_row(Vsl[dt * 16 + l16], half);
            o[dt] = __builtin_amdgcn_wmma_f32_16x16x32_bf16(
                false, pf, false, vf, (short)0, o[dt], false, false);
        }
        __syncthreads();
    }

    // finalize: divide by row sums, write [B,N,C] bf16 (C index = h*64 + d)
#pragma unroll
    for (int v = 0; v < 8; ++v) {
        float l = lrun[v];
        l += __shfl_xor(l, 1);
        l += __shfl_xor(l, 2);
        l += __shfl_xor(l, 4);
        l += __shfl_xor(l, 8);
        float inv = 1.0f / l;
        int row = b * SEQ_N + blockIdx.x * 128 + wave * 16 + v + 8 * half;
#pragma unroll
        for (int dt = 0; dt < 4; ++dt)
            Out[(size_t)row * WIDTH_C + h * HEAD_D + dt * 16 + l16] =
                f2bf(o[dt][v] * inv);
    }
}

// ---------------------------------------------------------------------------
// Host-side orchestration.
// Workspace budget ~126 MB (bf16 weights/activations + fp32 residual).
// ---------------------------------------------------------------------------
extern "C" void kernel_launch(void* const* d_in, const int* in_sizes, int n_in,
                              void* d_out, int out_size, void* d_ws, size_t ws_size,
                              hipStream_t stream) {
    (void)in_sizes; (void)n_in; (void)out_size; (void)ws_size;
    const float* x     = (const float*)d_in[0];
    const float* ln1g  = (const float*)d_in[1];
    const float* ln1b  = (const float*)d_in[2];
    const float* wqkv  = (const float*)d_in[3];
    const float* wproj = (const float*)d_in[4];
    const float* bproj = (const float*)d_in[5];
    const float* ln2g  = (const float*)d_in[6];
    const float* ln2b  = (const float*)d_in[7];
    const float* wfc1  = (const float*)d_in[8];
    const float* bfc1  = (const float*)d_in[9];
    const float* wfc2  = (const float*)d_in[10];
    const float* bfc2  = (const float*)d_in[11];
    float* out = (float*)d_out;

    char* ws = (char*)d_ws;
    size_t off = 0;
    auto take = [&](size_t bytes) -> void* {
        void* p = ws + off;
        off += (bytes + 255) & ~(size_t)255;
        return p;
    };
    const size_t E = ROWS_M * (size_t)WIDTH_C;              // 4M elements
    __bf16* wqkv_b  = (__bf16*)take((size_t)WIDTH_C * 3 * WIDTH_C * 2);
    __bf16* wproj_b = (__bf16*)take((size_t)WIDTH_C * WIDTH_C * 2);
    __bf16* wfc1_b  = (__bf16*)take((size_t)WIDTH_C * HIDDEN_F * 2);
    __bf16* wfc2_b  = (__bf16*)take((size_t)HIDDEN_F * WIDTH_C * 2);
    __bf16* h1      = (__bf16*)take(E * 2);
    __bf16* qb      = (__bf16*)take(E * 2);
    __bf16* kb      = (__bf16*)take(E * 2);
    __bf16* vb      = (__bf16*)take(E * 2);
    __bf16* attn_b  = (__bf16*)take(E * 2);
    float*  x1      = (float*)take(E * 4);
    __bf16* h2      = (__bf16*)take(E * 2);
    __bf16* h3      = (__bf16*)take((size_t)ROWS_M * HIDDEN_F * 2);

    // 1) cast weights to bf16
    {
        int n;
        n = WIDTH_C * 3 * WIDTH_C;
        cast_f32_bf16<<<(n + 255) / 256, 256, 0, stream>>>(wqkv, wqkv_b, n);
        n = WIDTH_C * WIDTH_C;
        cast_f32_bf16<<<(n + 255) / 256, 256, 0, stream>>>(wproj, wproj_b, n);
        n = WIDTH_C * HIDDEN_F;
        cast_f32_bf16<<<(n + 255) / 256, 256, 0, stream>>>(wfc1, wfc1_b, n);
        n = HIDDEN_F * WIDTH_C;
        cast_f32_bf16<<<(n + 255) / 256, 256, 0, stream>>>(wfc2, wfc2_b, n);
    }

    // 2) LN1 -> h1 (bf16)
    layernorm_bf16<<<ROWS_M, 256, 0, stream>>>(x, ln1g, ln1b, h1);

    // 3) QKV GEMM [4096,1024]x[1024,3072], scatter to q/k/v [B,H,N,D]
    gemm_bf16_wmma<0><<<dim3(3 * WIDTH_C / 128, ROWS_M / 128), 256, 0, stream>>>(
        h1, wqkv_b, ROWS_M, 3 * WIDTH_C, WIDTH_C,
        nullptr, nullptr, nullptr, nullptr, qb, kb, vb);

    // 4) attention -> attn_b (bf16, [B,N,C])
    attention_wmma<<<dim3(SEQ_N / 128, HEADS_H, BATCH_B), 256, 0, stream>>>(
        qb, kb, vb, attn_b);

    // 5) proj GEMM + bias + residual(x) -> x1 (f32)
    gemm_bf16_wmma<1><<<dim3(WIDTH_C / 128, ROWS_M / 128), 256, 0, stream>>>(
        attn_b, wproj_b, ROWS_M, WIDTH_C, WIDTH_C,
        bproj, x, x1, nullptr, nullptr, nullptr, nullptr);

    // 6) LN2 -> h2 (bf16)
    layernorm_bf16<<<ROWS_M, 256, 0, stream>>>(x1, ln2g, ln2b, h2);

    // 7) FC1 GEMM + bias + exact GELU -> h3 (bf16)
    gemm_bf16_wmma<2><<<dim3(HIDDEN_F / 128, ROWS_M / 128), 256, 0, stream>>>(
        h2, wfc1_b, ROWS_M, HIDDEN_F, WIDTH_C,
        bfc1, nullptr, nullptr, h3, nullptr, nullptr, nullptr);

    // 8) FC2 GEMM + bias + residual(x1) -> out (f32)
    gemm_bf16_wmma<1><<<dim3(WIDTH_C / 128, ROWS_M / 128), 256, 0, stream>>>(
        h3, wfc2_b, ROWS_M, WIDTH_C, HIDDEN_F,
        bfc2, x1, out, nullptr, nullptr, nullptr, nullptr);
}
